// AttentionBlock_987842478212
// MI455X (gfx1250) — matrix operand order repaired and checked by reading, not compiled, not run on hardware
//
#include <hip/hip_runtime.h>
#include <hip/hip_bf16.h>

// ---------------------------------------------------------------------------
// Problem constants (from reference)
#define NQ     16384
#define MCTX   4096
#define SIN    256
#define PDIM   256
#define YDIM   7
#define CHUNK  64      // KV rows per iteration in the attention kernel

typedef __attribute__((ext_vector_type(16))) __bf16 v16bf;
typedef __attribute__((ext_vector_type(8)))  __bf16 v8bf;
typedef __attribute__((ext_vector_type(8)))  float  v8f;
typedef __attribute__((ext_vector_type(4)))  float  v4f;
typedef unsigned int u32x4 __attribute__((ext_vector_type(4)));
typedef int          i32x4 __attribute__((ext_vector_type(4)));
typedef int          i32x8 __attribute__((ext_vector_type(8)));

#if __has_builtin(__builtin_amdgcn_tensor_load_to_lds)
#define HAVE_TDM 1
#else
#define HAVE_TDM 0
#endif

// ---- bf16 <-> f32: plain casts so the backend can use native cvt ops ------
static __device__ inline __bf16 f2bf(float f) { return (__bf16)f; }
static __device__ inline float  bf2f(__bf16 b) { return (float)b; }

// pack 8 floats (two v4f) into a v8bf
static __device__ inline v8bf cvt8(v4f a, v4f b) {
    v8bf r;
    r[0] = (__bf16)a[0]; r[1] = (__bf16)a[1]; r[2] = (__bf16)a[2]; r[3] = (__bf16)a[3];
    r[4] = (__bf16)b[0]; r[5] = (__bf16)b[1]; r[6] = (__bf16)b[2]; r[7] = (__bf16)b[3];
    return r;
}

// ---- LDS byte-offset of a __shared__ object (generic -> AS3 -> int) -------
typedef __attribute__((address_space(3))) const char* lds_cptr_t;
static __device__ inline unsigned lds_off(const void* p) {
    return (unsigned)(size_t)(lds_cptr_t)p;
}

// ---- WMMA wrapper ---------------------------------------------------------
static __device__ inline v8f wmma_bf16(v16bf a, v16bf b, v8f c) {
    // D = A(16x32 bf16) * B(32x16 bf16) + C(16x16 f32)
    return __builtin_amdgcn_wmma_f32_16x16x32_bf16(
        /*neg_a=*/false, a, /*neg_b=*/false, b,
        /*c_mod=*/(short)0, c, /*reuse_a=*/false, /*reuse_b=*/false);
}

// ---- Fragment loaders (ISA 7.12.2 layouts, wave32) ------------------------
// A-matrix 16x32 bf16: lane L -> row M = L&15, h = L>>4; lane's 16 halves are
// two contiguous 8-half runs at K = k0+h*8 and K = k0+16+h*8.
static __device__ inline v16bf load_a_frag(const __bf16* rowPtr, int k0, int h) {
    union { v16bf v; v8bf p[2]; } u;
    u.p[0] = *(const v8bf*)(rowPtr + k0 + h * 8);
    u.p[1] = *(const v8bf*)(rowPtr + k0 + 16 + h * 8);
    return u.v;
}
// B-matrix 32x16 bf16: lane L -> N = n0 + (L&15), h = L>>4; lane's 16 halves
// are K = k0+h*8..+8 and K = k0+16+h*8..+8 (same order as A).
// base is row-major [K][ldb] with N contiguous.
static __device__ inline v16bf load_b_frag(const __bf16* base, int ldb,
                                           int k0, int n0, int lane) {
    const int hb = lane >> 4;
    const __bf16* p = base + n0 + (lane & 15);
    v16bf r;
#pragma unroll
    for (int i = 0; i < 8; i++) {
        r[i]     = p[(size_t)(k0 + hb * 8 + i) * ldb];
        r[8 + i] = p[(size_t)(k0 + 16 + hb * 8 + i) * ldb];
    }
    return r;
}

#if HAVE_TDM
// ---- TDM: DMA a [rows x PDIM] bf16 tile (row-major, stride PDIM) to LDS ---
// D# built per CDNA5 ISA 8.3/8.4: group0 = {count/type/addrs}, group1 =
// {data_size, tensor dims/strides, tile dims}.  2-D tensor: groups 2/3 unused.
static __device__ inline void tdm_load_tile(const __bf16* gsrc,
                                            unsigned lds_byte_off, int rows) {
    const unsigned long long ga = (unsigned long long)(size_t)gsrc;
    u32x4 g0 = {
        1u,                                            // count=1, user mode
        lds_byte_off,                                  // lds_addr (bytes)
        (unsigned)(ga & 0xFFFFFFFFu),                  // global_addr[31:0]
        (unsigned)((ga >> 32) & 0x1FFFFFFu) | (2u << 30) // [56:32] | type=2
    };
    i32x8 g1 = {
        (int)(1u << 16),            // data_size=1 (2 bytes); wg_mask=0
        (int)((unsigned)PDIM << 16),// tensor_dim0[15:0] in [31:16]
        (int)((unsigned)MCTX << 16),// tensor_dim0 hi=0; tensor_dim1[15:0]
        (int)((unsigned)PDIM << 16),// tensor_dim1 hi=0; tile_dim0=PDIM
        rows,                       // tile_dim1=rows; tile_dim2=0
        PDIM,                       // tensor_dim0_stride[31:0]
        0, 0                        // stride hi / dim1_stride (2-D: unused)
    };
    i32x4 z4 = {};
    i32x8 z8 = {};
    __builtin_amdgcn_tensor_load_to_lds(g0, g1, z4, z4, z8, /*cpol=*/0);
}
#endif

// ---------------------------------------------------------------------------
// Kernel 0: convert Wq (f32 [SIN][PDIM]) -> bf16 row-major, once.
__global__ __launch_bounds__(256) void cvt_wq_kernel(
    const float* __restrict__ Wq, __bf16* __restrict__ WqB) {
    const size_t i = ((size_t)blockIdx.x * 256 + threadIdx.x) * 16;
    v4f a0 = *(const v4f*)(Wq + i);
    v4f a1 = *(const v4f*)(Wq + i + 4);
    v4f a2 = *(const v4f*)(Wq + i + 8);
    v4f a3 = *(const v4f*)(Wq + i + 12);
    *(v8bf*)(WqB + i)     = cvt8(a0, a1);
    *(v8bf*)(WqB + i + 8) = cvt8(a2, a3);
}

// ---------------------------------------------------------------------------
// Kernel 1: YK = y@Wk stored transposed [P][M] bf16 ; YV = y@Wv stored [M][P].
__global__ __launch_bounds__(256) void proj_y_kernel(
    const float* __restrict__ y, const float* __restrict__ Wk,
    const float* __restrict__ Wv, __bf16* __restrict__ YKT,
    __bf16* __restrict__ YV) {
    const int m = blockIdx.x;       // 0..MCTX-1
    const int p = threadIdx.x;      // 0..PDIM-1
    float yr[YDIM];
#pragma unroll
    for (int d = 0; d < YDIM; d++) yr[d] = y[m * YDIM + d];
    float k = 0.f, v = 0.f;
#pragma unroll
    for (int d = 0; d < YDIM; d++) {
        k = fmaf(yr[d], Wk[d * PDIM + p], k);
        v = fmaf(yr[d], Wv[d * PDIM + p], v);
    }
    YV[(size_t)m * PDIM + p]  = f2bf(v);
    YKT[(size_t)p * MCTX + m] = f2bf(k);
}

// ---------------------------------------------------------------------------
// Kernel 2: XQ = x @ Wq -> bf16 [NQ][PDIM].  Block = 16 query rows, 8 waves.
// The 16x256 x-tile is converted to bf16 cooperatively ONCE into LDS; waves
// pull A-fragments from LDS and B-fragments from the pre-converted bf16 Wq.
__global__ __launch_bounds__(256) void xq_kernel(
    const float* __restrict__ x, const __bf16* __restrict__ WqB,
    __bf16* __restrict__ XQ) {
    __shared__ __bf16 xTile[16][SIN];
    const int tid = threadIdx.x;
    const int lane = tid & 31, w = tid >> 5;
    const int h = lane >> 4, ln = lane & 15;
    const int q0 = blockIdx.x * 16;
    const int n0 = w * 32;

    // cooperative f32->bf16 staging: thread -> (row = tid>>4, 16 cols)
    {
        const int r  = tid >> 4;
        const int c0 = (tid & 15) * 16;
        const float* src = x + (size_t)(q0 + r) * SIN + c0;
        v4f a0 = *(const v4f*)(src);
        v4f a1 = *(const v4f*)(src + 4);
        v4f a2 = *(const v4f*)(src + 8);
        v4f a3 = *(const v4f*)(src + 12);
        *(v8bf*)&xTile[r][c0]     = cvt8(a0, a1);
        *(v8bf*)&xTile[r][c0 + 8] = cvt8(a2, a3);
    }
    __syncthreads();

    v8f acc0 = {}, acc1 = {};
#pragma unroll
    for (int kc = 0; kc < SIN / 32; kc++) {
        v16bf af = load_a_frag(&xTile[ln][0], kc * 32, h);
        v16bf b0 = load_b_frag(WqB, PDIM, kc * 32, n0, lane);
        v16bf b1 = load_b_frag(WqB, PDIM, kc * 32, n0 + 16, lane);
        acc0 = wmma_bf16(af, b0, acc0);
        acc1 = wmma_bf16(af, b1, acc1);
    }
    // C/D layout: VGPR v -> row = v + 8*h, col = lane&15
#pragma unroll
    for (int v = 0; v < 8; v++) {
        const int row = v + 8 * h;
        XQ[(size_t)(q0 + row) * PDIM + n0 + ln]      = f2bf(acc0[v]);
        XQ[(size_t)(q0 + row) * PDIM + n0 + 16 + ln] = f2bf(acc1[v]);
    }
}

// ---------------------------------------------------------------------------
// Kernel 3: fused attention.  Block = 32 queries, 256 threads (8 waves).
// Wave w -> (qt = w&1: which 16-query half, cg = w>>1: 16-col score group /
// 64-col output slab).  Per 64-KV chunk:
//   - wave 0 issues a TDM tensor_load_to_lds of the 64x256 V tile (overlapped
//     with the score phase), s_wait_tensorcnt before the barrier.
//   - score phase: 16x16 score tile (8 WMMA, K=256) -> exp/relu bf16 tiles.
//   - PV phase:    4 col tiles x 2 K-subchunks x {exp,relu} = 16 WMMA/wave,
//                  B-fragments from the TDM-staged LDS V tile.
//   - Z/R row sums via deterministic LDS partial reduction.
// Final: out = (0.1*S2 + S1/Z) / (0.1*R + 1).
__global__ __launch_bounds__(256) void attn_kernel(
    const __bf16* __restrict__ XQ, const __bf16* __restrict__ YKT,
    const __bf16* __restrict__ YV, float* __restrict__ out) {
    __shared__ __bf16 vTile[CHUNK][PDIM];    // 32 KB: TDM-staged V tile
    __shared__ __bf16 pTile[2][16][CHUNK];   // 4 KB: exp(score)
    __shared__ __bf16 rTile[2][16][CHUNK];   // 4 KB: relu(score)
    __shared__ float  Zpart[32][8], Rpart[32][8];
    __shared__ float  Zsh[32], Rsh[32];

    const int tid = threadIdx.x;
    const int lane = tid & 31, w = tid >> 5;
    const int h = lane >> 4, ln = lane & 15;
    const int qt = w & 1;                 // query half-tile (rows qt*16..+16)
    const int cg = w >> 1;                // score cols cg*16.. / out cols cg*64..
    const int q0 = blockIdx.x * 32;
    const int rrow = tid & 31, rseg = tid >> 5;   // reduction assignment

    if (tid < 32) { Zsh[tid] = 0.f; Rsh[tid] = 0.f; }

    // Preload this wave's Q fragments (16 rows x 256 K) -> 8 frags in VGPRs.
    v16bf qf[SIN / 32];
    const __bf16* qrow = XQ + (size_t)(q0 + qt * 16 + ln) * PDIM;
#pragma unroll
    for (int kc = 0; kc < SIN / 32; kc++) qf[kc] = load_a_frag(qrow, kc * 32, h);

    v8f aP[4] = {}, aR[4] = {};

#if HAVE_TDM
    const unsigned vtile_lds = lds_off(&vTile[0][0]);
#endif

    __syncthreads();

    for (int kv0 = 0; kv0 < MCTX; kv0 += CHUNK) {
        // ---- kick off async V-tile staging for this chunk ----
#if HAVE_TDM
        if (w == 0)
            tdm_load_tile(YV + (size_t)kv0 * PDIM, vtile_lds, CHUNK);
#else
        {
            const __bf16* src = YV + (size_t)kv0 * PDIM;
            __bf16* dst = &vTile[0][0];
            for (int i = tid * 8; i < CHUNK * PDIM; i += 256 * 8)
                *(v8bf*)(dst + i) = *(const v8bf*)(src + i);
        }
#endif

        // ---- score phase: s = (Q . K^T)/16 for cols [kv0+cg*16, +16) ----
        v8f s = {};
#pragma unroll
        for (int kc = 0; kc < SIN / 32; kc++) {
            v16bf b = load_b_frag(YKT, MCTX, kc * 32, kv0 + cg * 16, lane);
            s = wmma_bf16(qf[kc], b, s);
        }
#pragma unroll
        for (int v = 0; v < 8; v++) {
            const float sc = s[v] * 0.0625f;             // 1/sqrt(256)
            const float pe = __expf(sc);                 // fixed shift: |sc| is O(5)
            const float rr = sc > 0.f ? sc : 0.f;
            const int   row = v + 8 * h;
            pTile[qt][row][cg * 16 + ln] = f2bf(pe);
            rTile[qt][row][cg * 16 + ln] = f2bf(rr);
        }

#if HAVE_TDM
        if (w == 0) __builtin_amdgcn_s_wait_tensorcnt(0);
#endif
        __syncthreads();   // V tile + score tiles visible to everyone

        // ---- deterministic Z/R partial sums (thread: 8 cols of 1 row) ----
        {
            float zp = 0.f, zr = 0.f;
#pragma unroll
            for (int i = 0; i < 8; i++) {
                zp += bf2f(pTile[rrow >> 4][rrow & 15][rseg * 8 + i]);
                zr += bf2f(rTile[rrow >> 4][rrow & 15][rseg * 8 + i]);
            }
            Zpart[rrow][rseg] = zp;
            Rpart[rrow][rseg] = zr;
        }

        // ---- PV phase: S1 += exp(s).V ; S2 += relu(s).V  (LDS V tile) ----
#pragma unroll
        for (int kc2 = 0; kc2 < CHUNK; kc2 += 32) {
            v16bf ap = load_a_frag(&pTile[qt][ln][0], kc2, h);
            v16bf ar = load_a_frag(&rTile[qt][ln][0], kc2, h);
#pragma unroll
            for (int t = 0; t < 4; t++) {
                v16bf b = load_b_frag(&vTile[0][0], PDIM, kc2, cg * 64 + t * 16, lane);
                aP[t] = wmma_bf16(ap, b, aP[t]);
                aR[t] = wmma_bf16(ar, b, aR[t]);
            }
        }
        __syncthreads();   // PV reads done; partials visible

        // ---- fold partials into running Z/R (threads 0..31) ----
        if (tid < 32) {
            float z = 0.f, r = 0.f;
#pragma unroll
            for (int i = 0; i < 8; i++) { z += Zpart[tid][i]; r += Rpart[tid][i]; }
            Zsh[tid] += z;
            Rsh[tid] += r;
        }
        __syncthreads();
    }

    // ---- final combine: out = (0.1*S2 + S1/Z) / (0.1*R + 1) ----
#pragma unroll
    for (int t = 0; t < 4; t++) {
#pragma unroll
        for (int v = 0; v < 8; v++) {
            const int   row  = qt * 16 + v + 8 * h;
            const float invZ = 1.0f / Zsh[row];
            const float invD = 1.0f / (0.1f * Rsh[row] + 1.0f);
            const float o = (0.1f * aR[t][v] + aP[t][v] * invZ) * invD;
            out[(size_t)(q0 + row) * PDIM + cg * 64 + t * 16 + ln] = o;
        }
    }
}

// ---------------------------------------------------------------------------
extern "C" void kernel_launch(void* const* d_in, const int* in_sizes, int n_in,
                              void* d_out, int out_size, void* d_ws, size_t ws_size,
                              hipStream_t stream) {
    const float* x  = (const float*)d_in[0];
    const float* y  = (const float*)d_in[1];
    const float* Wq = (const float*)d_in[2];
    const float* Wk = (const float*)d_in[3];
    const float* Wv = (const float*)d_in[4];
    float* out = (float*)d_out;

    const size_t xq_bytes  = (size_t)NQ * PDIM * 2;     // 8 MB
    const size_t ykt_bytes = (size_t)PDIM * MCTX * 2;   // 2 MB
    const size_t yv_bytes  = (size_t)MCTX * PDIM * 2;   // 2 MB
    const size_t wq_bytes  = (size_t)SIN * PDIM * 2;    // 128 KB
    if (ws_size < xq_bytes + ykt_bytes + yv_bytes + wq_bytes) return;

    char* ws = (char*)d_ws;
    __bf16* XQ  = (__bf16*)ws;
    __bf16* YKT = (__bf16*)(ws + xq_bytes);
    __bf16* YV  = (__bf16*)(ws + xq_bytes + ykt_bytes);
    __bf16* WqB = (__bf16*)(ws + xq_bytes + ykt_bytes + yv_bytes);

    cvt_wq_kernel<<<(SIN * PDIM) / (256 * 16), 256, 0, stream>>>(Wq, WqB);
    proj_y_kernel<<<MCTX, 256, 0, stream>>>(y, Wk, Wv, YKT, YV);
    xq_kernel<<<NQ / 16, 256, 0, stream>>>(x, WqB, XQ);
    attn_kernel<<<NQ / 32, 256, 0, stream>>>(XQ, YKT, YV, out);
}
